// Transformer3DBlock_40252433498462
// MI455X (gfx1250) — matrix operand order repaired and checked by reading, not compiled, not run on hardware
//
#include <hip/hip_runtime.h>

// ---------------- constants ----------------
#define C_DIM   192
#define NHEADS  6
#define HD      32
#define N_WIN   144      // tokens per window (2*6*12)
#define TYPEWIN 60       // windows per lon slice (4*15)
#define NWINDOWS 900     // 15 * 60
#define TOKENS  129600   // 8*90*180
#define NHW     5400     // NWINDOWS * NHEADS
#define NMSQ    20736    // 144*144

typedef __attribute__((ext_vector_type(16))) __bf16 v16bf;
typedef __attribute__((ext_vector_type(8)))  float  v8f;

union FragB { unsigned u[8]; v16bf v; };

__device__ __forceinline__ unsigned short f2bf(float f) {
  unsigned u = __builtin_bit_cast(unsigned, f);
  u = (u + 0x7FFFu + ((u >> 16) & 1u)) >> 16;
  return (unsigned short)u;
}

__device__ __forceinline__ unsigned pack2bf(float lo, float hi) {
  return (unsigned)f2bf(lo) | ((unsigned)f2bf(hi) << 16);
}

// region id for the shifted-window mask (replicates _shift_window_mask)
__device__ __forceinline__ int region_cnt(int idx, int lon, int p, int q) {
  int z  = idx / 72, hh = (idx / 12) % 6, ww = idx % 12;
  int pl = p * 2 + z, la = q * 6 + hh, lo = lon * 12 + ww;
  int rp = (pl < 6)  ? 0 : ((pl < 7)  ? 1 : 2);
  int rl = (la < 84) ? 0 : ((la < 87) ? 1 : 2);
  int ro = (lo < 174)? 0 : ((lo < 183)? 1 : 2);
  return rp * 9 + rl * 3 + ro;
}

// ---------------- weight convert: W[K][N] f32 -> Wt[N][K] bf16 ----------------
__global__ void wconv_kernel(const float* __restrict__ W, unsigned short* __restrict__ Wt,
                             int K, int N) {
  int idx = blockIdx.x * 256 + threadIdx.x;
  if (idx >= K * N) return;
  int n = idx / K, k = idx % K;
  Wt[idx] = f2bf(W[(size_t)k * N + n]);
}

// ---------------- earth-position bias expansion (one-time, reused 15x) ----------------
// BiasF[(w*6+head)][n][m] = bias_table[EPI(n,m)][w][head]
__global__ void bias_expand_kernel(const float* __restrict__ btab, float* __restrict__ BiasF) {
  int idx = blockIdx.x * 256 + threadIdx.x;
  if (idx >= 360 * NMSQ) return;
  int wh = idx / NMSQ;
  int nm = idx % NMSQ;
  int n = nm / N_WIN, m2 = nm % N_WIN;
  int zi = n / 72,  hi = (n / 12) % 6,  wi = n % 12;
  int z2 = m2 / 72, h2 = (m2 / 12) % 6, w2 = m2 % 12;
  int epi = (zi + 2 * z2) * 828 + (hi + 6 * h2) * 23 + (wi - w2 + 11);
  BiasF[idx] = btab[(size_t)epi * 360 + wh];
}

// ---------------- LayerNorm (wave per row) ----------------
// PARTITION=1: output row r is in window order, source token via roll(-1,-3,-3)+partition
// PARTITION=0: identity row mapping (h -> H2)
template<int PARTITION>
__global__ __launch_bounds__(128) void ln_kernel(const float* __restrict__ x,
                                                 const float* __restrict__ g,
                                                 const float* __restrict__ b,
                                                 unsigned short* __restrict__ out) {
  int lane = threadIdx.x & 31;
  int wv   = threadIdx.x >> 5;
  int r    = blockIdx.x * 4 + wv;
  int token;
  if (PARTITION) {
    int win = r / N_WIN, n = r % N_WIN;
    int lon = win / TYPEWIN, w = win % TYPEWIN;
    int p = w / 15, q = w % 15;
    int zp = n / 72, zl = (n / 12) % 6, zw = n % 12;
    int pl = p * 2 + zp, la = q * 6 + zl, lo = lon * 12 + zw;
    int spl = (pl + 1) & 7;
    int sla = (la + 3) % 90;
    int slo = (lo + 3) % 180;   // reference rolls lon axis by s_lat=3 (faithful)
    token = (spl * 90 + sla) * 180 + slo;
  } else {
    token = r;
  }
  const float* xr = x + (size_t)token * C_DIM;
  float v[6]; float s = 0.f, s2 = 0.f;
#pragma unroll
  for (int j = 0; j < 6; ++j) {
    v[j] = xr[j * 32 + lane];
    s += v[j]; s2 += v[j] * v[j];
  }
#pragma unroll
  for (int m = 1; m < 32; m <<= 1) {
    s  += __shfl_xor(s,  m, 32);
    s2 += __shfl_xor(s2, m, 32);
  }
  float mean = s * (1.f / 192.f);
  float var  = s2 * (1.f / 192.f) - mean * mean;
  float inv  = rsqrtf(var + 1e-5f);
  unsigned short* orow = out + (size_t)r * C_DIM;
#pragma unroll
  for (int j = 0; j < 6; ++j) {
    int c = j * 32 + lane;
    orow[c] = f2bf((v[j] - mean) * inv * g[c] + b[c]);
  }
}

// ---------------- generic bf16 WMMA GEMM, wave = 32x64 tile ----------------
// A: [M x K] bf16 row-major; Bt: [N x K] bf16 (i.e., B transposed)
// 2 waves per block -> block tile 64 rows x 64 cols. Each B fragment feeds 2 WMMAs.
// MODE 0: QKV  -> scatter Q(scaled)/K/Vt bf16
// MODE 1: proj -> window-reverse + roll(+1,+3,+6) + residual, f32 out
// MODE 2: FC1  -> GELU, bf16 out [M x 768]
// MODE 3: FC2  -> residual, f32 out [M x 192]
template<int MODE>
__global__ __launch_bounds__(64) void gemm_wmma(
    const unsigned short* __restrict__ A, const unsigned short* __restrict__ Bt,
    const float* __restrict__ bias, int K,
    unsigned short* __restrict__ outQ, unsigned short* __restrict__ outK,
    unsigned short* __restrict__ outVt,
    const float* __restrict__ shortcut, float* __restrict__ outF,
    unsigned short* __restrict__ outB) {
  int lane = threadIdx.x & 31;
  int wave = threadIdx.x >> 5;        // 0..1
  int lan  = lane & 15;
  int hk   = lane >> 4;
  int rowBase = blockIdx.y * 64 + wave * 32;
  int colBase = blockIdx.x * 64;

  v8f acc[2][4] = {};
  for (int kt = 0; kt < K; kt += 32) {
    FragB a[2];
#pragma unroll
    for (int r = 0; r < 2; ++r) {
      const unsigned short* arow = A + (size_t)(rowBase + r * 16 + lan) * K + kt;
#pragma unroll
      for (int v = 0; v < 8; ++v) {
        int k0 = hk * 16 + (v << 1);
        a[r].u[v] = *(const unsigned*)(arow + k0);
      }
      if (kt + 32 < K) __builtin_prefetch(arow + 32, 0, 3);   // global_prefetch_b8
    }
#pragma unroll
    for (int nt = 0; nt < 4; ++nt) {
      int n = colBase + nt * 16 + lan;
      const unsigned short* brow = Bt + (size_t)n * K + kt + hk * 16;
      FragB bf;
#pragma unroll
      for (int v = 0; v < 8; ++v) bf.u[v] = *(const unsigned*)(brow + v * 2);
      acc[0][nt] = __builtin_amdgcn_wmma_f32_16x16x32_bf16(
          false, a[0].v, false, bf.v, (short)0, acc[0][nt], false, false);
      acc[1][nt] = __builtin_amdgcn_wmma_f32_16x16x32_bf16(
          false, a[1].v, false, bf.v, (short)0, acc[1][nt], false, false);
    }
  }

  // ---------------- epilogue ----------------
  // C-layout: VGPR i -> row (i + hk*8), lane low nibble -> col.
  // Hoist all column(nt)-invariant math out of the row loops.
  int   cA[4]; float bA[4];
  int   qiA[4], pcA[4];            // MODE 0: qkv-index and per-column address part
#pragma unroll
  for (int nt = 0; nt < 4; ++nt) {
    int c = colBase + nt * 16 + lan;
    cA[nt] = c;
    bA[nt] = bias[c];
    if (MODE == 0) {
      int qi = c / C_DIM, rem = c % C_DIM;
      int head = rem >> 5, d = rem & 31;
      qiA[nt] = qi;
      // addr(Q/K) = win*27648 + head*4608 + n*32 + d
      // addr(Vt)  = win*27648 + head*4608 + d*144 + n
      pcA[nt] = (qi == 2) ? (head * 4608 + d * N_WIN) : (head * 4608 + d);
    }
  }

#pragma unroll
  for (int r = 0; r < 2; ++r) {
#pragma unroll
    for (int i = 0; i < 8; ++i) {
      int row = rowBase + r * 16 + i + hk * 8;
      if (MODE == 0) {
        int win = row / N_WIN, n = row % N_WIN;
        int wbase = win * 27648;
#pragma unroll
        for (int nt = 0; nt < 4; ++nt) {
          float val = acc[r][nt][i] + bA[nt];
          if (qiA[nt] == 0)      outQ[wbase + pcA[nt] + n * HD] = f2bf(val * 0.17677669529663687f);
          else if (qiA[nt] == 1) outK[wbase + pcA[nt] + n * HD] = f2bf(val);
          else                   outVt[wbase + pcA[nt] + n]     = f2bf(val);
        }
      } else if (MODE == 1) {
        int win = row / N_WIN, n = row % N_WIN;
        int lon = win / TYPEWIN, w = win % TYPEWIN;
        int p = w / 15, q = w % 15;
        int zp = n / 72, zl = (n / 12) % 6, zw = n % 12;
        int pl = p * 2 + zp, la = q * 6 + zl, lo = lon * 12 + zw;
        int tpl = (pl + 1) & 7, tla = (la + 3) % 90, tlo = (lo + 6) % 180;
        size_t tokBase = (((size_t)tpl * 90 + tla) * 180 + tlo) * C_DIM;
#pragma unroll
        for (int nt = 0; nt < 4; ++nt) {
          float val = acc[r][nt][i] + bA[nt];
          outF[tokBase + cA[nt]] = shortcut[tokBase + cA[nt]] + val;
        }
      } else if (MODE == 2) {
        size_t rb = (size_t)row * 768;
#pragma unroll
        for (int nt = 0; nt < 4; ++nt) {
          float val = acc[r][nt][i] + bA[nt];
          float gl = 0.5f * val * (1.0f + erff(val * 0.70710678118654752f));
          outB[rb + cA[nt]] = f2bf(gl);
        }
      } else {
        size_t rb = (size_t)row * C_DIM;
#pragma unroll
        for (int nt = 0; nt < 4; ++nt) {
          float val = acc[r][nt][i] + bA[nt];
          outF[rb + cA[nt]] = shortcut[rb + cA[nt]] + val;
        }
      }
    }
  }
}

// ---------------- attention: one block per (window, head), 9 waves ----------------
__global__ __launch_bounds__(288) void attn_kernel(
    const unsigned short* __restrict__ Qg, const unsigned short* __restrict__ Kg,
    const unsigned short* __restrict__ Vtg, const float* __restrict__ BiasF,
    unsigned short* __restrict__ Og) {
  __shared__ __attribute__((aligned(16))) unsigned short Vt_lds[32 * 160];
  __shared__ __attribute__((aligned(16))) unsigned short P_lds[9 * 16 * 160];

  int hw = blockIdx.x;                  // 0..5399
  int win = hw / NHEADS, head = hw % NHEADS;
  int lon = win / TYPEWIN, w = win % TYPEWIN;
  int p = w / 15, q = w % 15;
  int tid = threadIdx.x;
  int lane = tid & 31, wave = tid >> 5; // wave = row strip 0..8
  int lan = lane & 15, hk = lane >> 4;

  // stage V^T [32 x 144] -> LDS [32 x 160], zero-padded K
  const unsigned short* vsrc = Vtg + (size_t)hw * 32 * N_WIN;
  for (int idx = tid; idx < 32 * 80; idx += 288) {
    int n = idx / 80, k = (idx % 80) * 2;
    unsigned val = (k < N_WIN) ? *(const unsigned*)(vsrc + n * N_WIN + k) : 0u;
    *(unsigned*)(Vt_lds + n * 160 + k) = val;
  }
  __syncthreads();

  // Q fragment (A-layout, single k-step: hd = 32)
  FragB qa;
  const unsigned short* qrow = Qg + ((size_t)hw * N_WIN + wave * 16 + lan) * HD;
#pragma unroll
  for (int v = 0; v < 8; ++v) {
    int k0 = hk * 16 + (v << 1);
    qa.u[v] = *(const unsigned*)(qrow + k0);
  }

  // S = Q * K^T : 9 column tiles
  v8f s[9];
#pragma unroll
  for (int ct = 0; ct < 9; ++ct) {
    FragB kb;
    const unsigned short* krow = Kg + ((size_t)hw * N_WIN + ct * 16 + lan) * HD + hk * 16;
#pragma unroll
    for (int v = 0; v < 8; ++v) kb.u[v] = *(const unsigned*)(krow + v * 2);
    v8f z = {};
    s[ct] = __builtin_amdgcn_wmma_f32_16x16x32_bf16(
        false, qa.v, false, kb.v, (short)0, z, false, false);
  }

  // precomputed earth-position bias (coalesced) + analytic shift-window mask
  int rcnt[8], rid[8];
#pragma unroll
  for (int i = 0; i < 8; ++i) {
    int nr = wave * 16 + i + hk * 8;
    rid[i] = nr;
    rcnt[i] = region_cnt(nr, lon, p, q);
  }
  const float* brow = BiasF + (size_t)(w * NHEADS + head) * NMSQ;
#pragma unroll
  for (int ct = 0; ct < 9; ++ct) {
    int nc = ct * 16 + lan;
    int ccnt = region_cnt(nc, lon, p, q);
#pragma unroll
    for (int i = 0; i < 8; ++i) {
      float bv = brow[rid[i] * N_WIN + nc];
      s[ct][i] += bv + ((rcnt[i] != ccnt) ? -100.0f : 0.0f);
    }
  }

  // row-wise softmax (rows live in 16-lane groups; xor 1/2/4/8 stays in group)
#pragma unroll
  for (int i = 0; i < 8; ++i) {
    float mx = -3.0e38f;
#pragma unroll
    for (int ct = 0; ct < 9; ++ct) mx = fmaxf(mx, s[ct][i]);
    mx = fmaxf(mx, __shfl_xor(mx, 1, 32));
    mx = fmaxf(mx, __shfl_xor(mx, 2, 32));
    mx = fmaxf(mx, __shfl_xor(mx, 4, 32));
    mx = fmaxf(mx, __shfl_xor(mx, 8, 32));
    float sm = 0.f;
#pragma unroll
    for (int ct = 0; ct < 9; ++ct) { float e = __expf(s[ct][i] - mx); s[ct][i] = e; sm += e; }
    sm += __shfl_xor(sm, 1, 32);
    sm += __shfl_xor(sm, 2, 32);
    sm += __shfl_xor(sm, 4, 32);
    sm += __shfl_xor(sm, 8, 32);
    float inv = 1.0f / sm;
#pragma unroll
    for (int ct = 0; ct < 9; ++ct) s[ct][i] *= inv;
  }

  // P -> LDS (bf16, K padded to 160)
  unsigned short* Pw = P_lds + wave * 16 * 160;
#pragma unroll
  for (int ct = 0; ct < 9; ++ct)
#pragma unroll
    for (int i = 0; i < 8; ++i)
      Pw[(i + hk * 8) * 160 + ct * 16 + lan] = f2bf(s[ct][i]);
#pragma unroll
  for (int i = 0; i < 8; ++i) Pw[(i + hk * 8) * 160 + 144 + lan] = 0;
  __syncthreads();

  // O = P x V  (K = 160, N = 32 -> two 16-col tiles)
  v8f o0 = {}, o1 = {};
  for (int kt = 0; kt < 160; kt += 32) {
    FragB pa;
    const unsigned short* prow = Pw + lan * 160 + kt;
#pragma unroll
    for (int v = 0; v < 8; ++v) {
      int k0 = hk * 16 + (v << 1);
      pa.u[v] = *(const unsigned*)(prow + k0);
    }
    FragB vb0, vb1;
    const unsigned short* v0row = Vt_lds + lan * 160 + kt + hk * 16;
    const unsigned short* v1row = Vt_lds + (16 + lan) * 160 + kt + hk * 16;
#pragma unroll
    for (int v = 0; v < 8; ++v) {
      vb0.u[v] = *(const unsigned*)(v0row + v * 2);
      vb1.u[v] = *(const unsigned*)(v1row + v * 2);
    }
    o0 = __builtin_amdgcn_wmma_f32_16x16x32_bf16(false, pa.v, false, vb0.v, (short)0, o0, false, false);
    o1 = __builtin_amdgcn_wmma_f32_16x16x32_bf16(false, pa.v, false, vb1.v, (short)0, o1, false, false);
  }

  // write O in window-row order for the proj GEMM: [win*144+n][head*32 + d]
#pragma unroll
  for (int i = 0; i < 8; ++i) {
    int n = wave * 16 + i + hk * 8;
    size_t base = ((size_t)win * N_WIN + n) * C_DIM + head * HD;
    Og[base + lan]      = f2bf(o0[i]);
    Og[base + 16 + lan] = f2bf(o1[i]);
  }
}

// ---------------- host-side orchestration ----------------
extern "C" void kernel_launch(void* const* d_in, const int* in_sizes, int n_in,
                              void* d_out, int out_size, void* d_ws, size_t ws_size,
                              hipStream_t stream) {
  const float* x       = (const float*)d_in[0];
  const float* n1g     = (const float*)d_in[1];
  const float* n1b     = (const float*)d_in[2];
  const float* qkv_w   = (const float*)d_in[3];
  const float* qkv_b   = (const float*)d_in[4];
  const float* btab    = (const float*)d_in[5];
  const float* proj_w  = (const float*)d_in[6];
  const float* proj_b  = (const float*)d_in[7];
  const float* n2g     = (const float*)d_in[8];
  const float* n2b     = (const float*)d_in[9];
  const float* fc1_w   = (const float*)d_in[10];
  const float* fc1_b   = (const float*)d_in[11];
  const float* fc2_w   = (const float*)d_in[12];
  const float* fc2_b   = (const float*)d_in[13];
  float* out = (float*)d_out;

  // workspace layout (all offsets 4B-aligned)
  unsigned short* Wqkv  = (unsigned short*)d_ws;            // 576*192
  unsigned short* Wproj = Wqkv  + 110592;                   // 192*192
  unsigned short* Wfc1  = Wproj + 36864;                    // 768*192
  unsigned short* Wfc2  = Wfc1  + 147456;                   // 192*768
  unsigned short* BIG   = Wfc2  + 147456;
  const size_t SEG = 24883200;                              // 129600*192 elems
  unsigned short* XW  = BIG;                                // LN1 output / reused as O
  unsigned short* Qb  = BIG + SEG;
  unsigned short* Kb  = BIG + 2 * SEG;
  unsigned short* Vtb = BIG + 3 * SEG;
  unsigned short* G   = BIG;                                // GELU out aliases XW..Vt (dead)
  float*          h   = (float*)(BIG + 4 * SEG);            // residual stream, f32
  unsigned short* H2  = (unsigned short*)(h + SEG);         // LN2 output
  float*          BiasF = (float*)(H2 + SEG);               // 360*20736 f32 (~30 MB)
  unsigned short* O   = XW;

  // 1) convert + transpose weights to bf16; expand earth-position bias
  wconv_kernel<<<(110592 + 255) / 256, 256, 0, stream>>>(qkv_w,  Wqkv, 192, 576);
  wconv_kernel<<<( 36864 + 255) / 256, 256, 0, stream>>>(proj_w, Wproj, 192, 192);
  wconv_kernel<<<(147456 + 255) / 256, 256, 0, stream>>>(fc1_w,  Wfc1, 192, 768);
  wconv_kernel<<<(147456 + 255) / 256, 256, 0, stream>>>(fc2_w,  Wfc2, 768, 192);
  bias_expand_kernel<<<(360 * NMSQ + 255) / 256, 256, 0, stream>>>(btab, BiasF);

  // 2) LN1 + roll + window partition -> XW (bf16, window order)
  ln_kernel<1><<<TOKENS / 4, 128, 0, stream>>>(x, n1g, n1b, XW);

  // 3) QKV GEMM (129600x576, K=192) -> Q(scaled)/K/Vt
  gemm_wmma<0><<<dim3(576 / 64, TOKENS / 64), 64, 0, stream>>>(
      XW, Wqkv, qkv_b, 192, Qb, Kb, Vtb, nullptr, nullptr, nullptr);

  // 4) windowed attention with bias + mask + softmax -> O (reuses XW)
  attn_kernel<<<NHW, 288, 0, stream>>>(Qb, Kb, Vtb, BiasF, O);

  // 5) proj GEMM + window reverse + roll + residual -> h (f32)
  gemm_wmma<1><<<dim3(192 / 64, TOKENS / 64), 64, 0, stream>>>(
      O, Wproj, proj_b, 192, nullptr, nullptr, nullptr, x, h, nullptr);

  // 6) LN2 -> H2 (bf16)
  ln_kernel<0><<<TOKENS / 4, 128, 0, stream>>>(h, n2g, n2b, H2);

  // 7) FC1 + GELU -> G (bf16, aliases dead XW/Q/K/Vt)
  gemm_wmma<2><<<dim3(768 / 64, TOKENS / 64), 64, 0, stream>>>(
      H2, Wfc1, fc1_b, 192, nullptr, nullptr, nullptr, nullptr, nullptr, G);

  // 8) FC2 + residual -> out (f32)
  gemm_wmma<3><<<dim3(192 / 64, TOKENS / 64), 64, 0, stream>>>(
      G, Wfc2, fc2_b, 768, nullptr, nullptr, nullptr, h, out, nullptr);
}
